// CausalLSTMNodeCell_16432544874516
// MI455X (gfx1250) — matrix operand order repaired and checked by reading, not compiled, not run on hardware
//
#include <hip/hip_runtime.h>

typedef __attribute__((ext_vector_type(16))) _Float16 v16h;
typedef __attribute__((ext_vector_type(4)))  _Float16 v4h;
typedef __attribute__((ext_vector_type(8)))  float    v8f;

#define BLK 256
#define WAVES 8          // 256 threads / wave32
#define HD 16            // hidden dim
#define DD 32            // input dim
#define NTILE 9          // 9 gate blocks of 16 cols: i f o a r0 r1 r2 n1 n2

// ws layout:
//   bytes [0 .. 18432): 18 weight fragments (f16, per-lane WMMA B order)
//                       h-side tiles carry combined bias in their K=16 row
//   bytes [65536 .. 65536 + B*16*6): sr = sigmoid(child_r) as f16, flat [B*48]
//   next  B*16*2 bytes:              g1 = sigmoid(n1) as f16, flat [B*16]
#define FRAG_HALVES (18 * 512)        // 18 tiles * 32 lanes * 16 halves
#define SMEM_DW (FRAG_HALVES / 2)     // 4608 dwords = 18432 B

__device__ __forceinline__ float fsig(float x) {
  // sigmoid(x) = 1 / (1 + 2^(-x*log2e))  -> v_exp_f32 + v_rcp_f32
  float e = __builtin_amdgcn_exp2f(-1.4426950408889634f * x);
  return __builtin_amdgcn_rcpf(1.0f + e);
}
__device__ __forceinline__ float ftanh(float x) {
  // tanh(x) = 1 - 2/(2^(2x*log2e) + 1)
  float e = __builtin_amdgcn_exp2f(2.8853900817779268f * x);
  return 1.0f - 2.0f * __builtin_amdgcn_rcpf(e + 1.0f);
}

// ---------------------------------------------------------------------------
// Prep: convert f32 weights into f16 WMMA B-fragments (per-lane order).
// B-matrix 32x16 f16, contiguous k order used here (wave32): lane L holds column N=L%16;
// lanes 0-15 hold K=0..15 (16 halves), lanes 16-31 hold K=16..31.
// Tiles 0..8  = x-weights (Ksrc=32).
// Tiles 9..17 = h-weights (Ksrc=16); K=16 row carries the combined bias
//               (bx+bh), matching A_h[K=16]=1.0 in the main kernel, so the
//               bias accumulates for free inside the h-side WMMA.
// ---------------------------------------------------------------------------
__global__ __launch_bounds__(BLK) void prep_kernel(
    const float* Wifo_x, const float* Wifo_h,
    const float* Wa_x,   const float* Wa_h,
    const float* Wr_x,   const float* Wr_h,
    const float* Wn1_x,  const float* Wn1_h,
    const float* Wn2_x,  const float* Wn2_h,
    const float* bifo_x, const float* bifo_h,
    const float* ba_x,   const float* ba_h,
    const float* br_x,   const float* br_h,
    const float* bn1_x,  const float* bn1_h,
    const float* bn2_x,  const float* bn2_h,
    void* ws)
{
  int t = blockIdx.x * BLK + threadIdx.x;

  const float* WX[NTILE] = {Wifo_x, Wifo_x, Wifo_x, Wa_x, Wr_x, Wr_x, Wr_x, Wn1_x, Wn2_x};
  const float* WH[NTILE] = {Wifo_h, Wifo_h, Wifo_h, Wa_h, Wr_h, Wr_h, Wr_h, Wn1_h, Wn2_h};
  const float* BX[NTILE] = {bifo_x, bifo_x, bifo_x, ba_x, br_x, br_x, br_x, bn1_x, bn2_x};
  const float* BH[NTILE] = {bifo_h, bifo_h, bifo_h, ba_h, br_h, br_h, br_h, bn1_h, bn2_h};
  const int    NC[NTILE] = {48, 48, 48, 16, 48, 48, 48, 16, 16};
  const int    CO[NTILE] = {0, 16, 32, 0, 0, 16, 32, 0, 0};

  if (t < 18 * 32) {
    int tile = t >> 5, lane = t & 31;
    bool is_x = (tile < NTILE);
    int j     = is_x ? tile : tile - NTILE;
    int Ksrc  = is_x ? 32 : 16;
    const float* W = is_x ? WX[j] : WH[j];
    int nc = NC[j], co = CO[j];
    int n  = lane & 15;
    int kb = (lane >= 16) ? 16 : 0;
    _Float16* outp = (_Float16*)ws + (size_t)tile * 512 + (size_t)lane * 16;
#pragma unroll
    for (int e = 0; e < 16; ++e) {
      int k = kb + e;
      float v;
      if (k < Ksrc)                     v = W[k * nc + co + n];
      else if (!is_x && k == 16)        v = BX[j][co + n] + BH[j][co + n];  // bias row
      else                              v = 0.0f;
      outp[e] = (_Float16)v;
    }
  }
}

// ---------------------------------------------------------------------------
// Kernel 1: per 16-row tile, 18 WMMAs (x-side K=32; h-side K=16 + bias at
// K=16 via A_h=1.0), then fused gate math. Writes h, c, sig(n2)*h (stashed in
// the n-output slot), g1 (f16, ws), sr = sigmoid(child_r) (f16 flat [B*48]).
// ---------------------------------------------------------------------------
__global__ __launch_bounds__(BLK) void cell_kernel(
    const float* __restrict__ x, const float* __restrict__ hp,
    const float* __restrict__ cp, const unsigned int* __restrict__ wsu,
    void* __restrict__ srv, void* __restrict__ g1v,
    float* __restrict__ out, int Btot)
{
  __shared__ __align__(16) unsigned int smem[SMEM_DW];
  for (int i = threadIdx.x; i < SMEM_DW; i += BLK) smem[i] = wsu[i];
  __syncthreads();

  int lane = threadIdx.x & 31;
  int wid  = threadIdx.x >> 5;
  int m0   = (blockIdx.x * WAVES + wid) * 16;
  if (m0 >= Btot) return;

  int n   = lane & 15;
  int hi  = lane >> 4;        // 0: lanes 0-15, 1: lanes 16-31
  int kbA = hi ? 16 : 0;      // A 16x32 f16, contiguous k order used here: lanes<16 K{0..15}, lanes>=16 K{16..31}
  int rbase = m0 + hi * 8;    // C/D 16x16 f32: VGPR k -> row rbase+k, col n

  // Build A fragments (x: full K=32; h: K=16 valid, K=16 slot = 1.0 for bias)
  const float* xr = x  + (size_t)(m0 + n) * DD;
  const float* hr = hp + (size_t)(m0 + n) * HD;
  v16h xa, ha;
#pragma unroll
  for (int e = 0; e < 8; ++e) {
    xa[e]     = (_Float16)xr[kbA + e];
    xa[e + 8] = (_Float16)xr[kbA + 8 + e];
    ha[e]     = hi ? (_Float16)0.0f : (_Float16)hr[e];
    ha[e + 8] = hi ? (_Float16)0.0f : (_Float16)hr[8 + e];
  }
  if (hi == 1) ha[0] = (_Float16)1.0f;   // A_h[m][16] = 1.0 -> picks up bias row of Wh

  // Early c_prev loads: latency hides under the 18 WMMAs below.
  float cprev[8];
#pragma unroll
  for (int k = 0; k < 8; ++k) cprev[k] = cp[(long)(rbase + k) * HD + n];

  union FragU { uint4 q[2]; v16h h; };
  v8f zc = {0.f, 0.f, 0.f, 0.f, 0.f, 0.f, 0.f, 0.f};
  v8f acc[NTILE];

#pragma unroll
  for (int j = 0; j < NTILE; ++j) {
    FragU wx, wh;
    int bx = j * 256 + lane * 8;
    int bh = (NTILE + j) * 256 + lane * 8;
    wx.q[0] = *(const uint4*)&smem[bx];
    wx.q[1] = *(const uint4*)&smem[bx + 4];
    wh.q[0] = *(const uint4*)&smem[bh];
    wh.q[1] = *(const uint4*)&smem[bh + 4];

    v8f c = __builtin_amdgcn_wmma_f32_16x16x32_f16(false, ha, false, wh.h, (short)0, zc, false, false);
    c     = __builtin_amdgcn_wmma_f32_16x16x32_f16(false, xa, false, wx.h, (short)0, c,  false, false);
    acc[j] = c;
  }

  long BHp = (long)Btot * HD;
  _Float16* srh = (_Float16*)srv;
  _Float16* g1h = (_Float16*)g1v;

#pragma unroll
  for (int k = 0; k < 8; ++k) {
    int row = rbase + k;
    long idx = (long)row * HD + n;

    float iv = fsig(acc[0][k]);
    float fv = fsig(acc[1][k]);
    float ov = fsig(acc[2][k]);
    float av = ftanh(acc[3][k]);
    float cnew = iv * av + fv * cprev[k];
    float hnew = ov * ftanh(cnew);

    // sr = sigmoid(child_r), stored flat row-major [B,48] for kernel 2
    long b48 = (long)row * 48 + n;
    srh[b48]      = (_Float16)fsig(acc[4][k]);
    srh[b48 + 16] = (_Float16)fsig(acc[5][k]);
    srh[b48 + 32] = (_Float16)fsig(acc[6][k]);

    g1h[idx] = (_Float16)fsig(acc[7][k]);

    out[idx]           = fsig(acc[8][k]) * hnew;  // stash sig(n2)*h in n-slot
    out[BHp + idx]     = hnew;                    // output h
    out[2 * BHp + idx] = cnew;                    // output c
  }
}

// ---------------------------------------------------------------------------
// Kernel 2: TF row-major reshape scramble collapses to a flat streaming pass:
// n[i] = g1[i] * sum_c sr_flat[c*BH + i] * child[c*BH + i] + stash[i]
// ---------------------------------------------------------------------------
__global__ __launch_bounds__(BLK) void vertical_kernel(
    const void* __restrict__ srv, const void* __restrict__ g1v,
    const float* __restrict__ child, float* __restrict__ out_n, long BHp)
{
  long i = ((long)blockIdx.x * BLK + threadIdx.x) * 4;
  if (i >= BHp) return;

  const _Float16* srh = (const _Float16*)srv;
  const _Float16* g1h = (const _Float16*)g1v;

  float4 ch0 = *(const float4*)(child + i);
  float4 ch1 = *(const float4*)(child + BHp + i);
  float4 ch2 = *(const float4*)(child + 2 * BHp + i);
  v4h s0 = *(const v4h*)(srh + i);
  v4h s1 = *(const v4h*)(srh + BHp + i);
  v4h s2 = *(const v4h*)(srh + 2 * BHp + i);
  v4h g  = *(const v4h*)(g1h + i);
  float4 stash = *(const float4*)(out_n + i);

  float4 r;
  r.x = (float)g[0] * ((float)s0[0] * ch0.x + (float)s1[0] * ch1.x + (float)s2[0] * ch2.x) + stash.x;
  r.y = (float)g[1] * ((float)s0[1] * ch0.y + (float)s1[1] * ch1.y + (float)s2[1] * ch2.y) + stash.y;
  r.z = (float)g[2] * ((float)s0[2] * ch0.z + (float)s1[2] * ch1.z + (float)s2[2] * ch2.z) + stash.z;
  r.w = (float)g[3] * ((float)s0[3] * ch0.w + (float)s1[3] * ch1.w + (float)s2[3] * ch2.w) + stash.w;
  *(float4*)(out_n + i) = r;
}

extern "C" void kernel_launch(void* const* d_in, const int* in_sizes, int n_in,
                              void* d_out, int out_size, void* d_ws, size_t ws_size,
                              hipStream_t stream) {
  const float* x       = (const float*)d_in[0];
  const float* h_prev  = (const float*)d_in[1];
  const float* c_prev  = (const float*)d_in[2];
  const float* child_n = (const float*)d_in[3];
  const float* W_ifo_x = (const float*)d_in[4];
  const float* b_ifo_x = (const float*)d_in[5];
  const float* W_ifo_h = (const float*)d_in[6];
  const float* b_ifo_h = (const float*)d_in[7];
  const float* W_a_x   = (const float*)d_in[8];
  const float* b_a_x   = (const float*)d_in[9];
  const float* W_a_h   = (const float*)d_in[10];
  const float* b_a_h   = (const float*)d_in[11];
  const float* W_r_x   = (const float*)d_in[12];
  const float* b_r_x   = (const float*)d_in[13];
  const float* W_r_h   = (const float*)d_in[14];
  const float* b_r_h   = (const float*)d_in[15];
  const float* W_n1_x  = (const float*)d_in[16];
  const float* b_n1_x  = (const float*)d_in[17];
  const float* W_n1_h  = (const float*)d_in[18];
  const float* b_n1_h  = (const float*)d_in[19];
  const float* W_n2_x  = (const float*)d_in[20];
  const float* b_n2_x  = (const float*)d_in[21];
  const float* W_n2_h  = (const float*)d_in[22];
  const float* b_n2_h  = (const float*)d_in[23];

  int B = in_sizes[0] / DD;
  size_t BHp = (size_t)B * HD;

  prep_kernel<<<3, BLK, 0, stream>>>(
      W_ifo_x, W_ifo_h, W_a_x, W_a_h, W_r_x, W_r_h, W_n1_x, W_n1_h, W_n2_x, W_n2_h,
      b_ifo_x, b_ifo_h, b_a_x, b_a_h, b_r_x, b_r_h, b_n1_x, b_n1_h, b_n2_x, b_n2_h,
      d_ws);

  char* wsb = (char*)d_ws;
  void* srv = (void*)(wsb + 65536);                 // f16 [B*48]
  void* g1v = (void*)(wsb + 65536 + BHp * 6);       // f16 [B*16]

  int nblk = (B + (WAVES * 16) - 1) / (WAVES * 16);
  cell_kernel<<<nblk, BLK, 0, stream>>>(x, h_prev, c_prev,
                                        (const unsigned int*)d_ws,
                                        srv, g1v, (float*)d_out, B);

  long nq = (long)((BHp + 3) / 4);
  int gb = (int)((nq + BLK - 1) / BLK);
  vertical_kernel<<<gb, BLK, 0, stream>>>(srv, g1v, child_n, (float*)d_out, (long)BHp);
}
